// SRUpp_71966472012537
// MI455X (gfx1250) — matrix operand order repaired and checked by reading, not compiled, not run on hardware
//
#include <hip/hip_runtime.h>

// ---------------- problem constants ----------------
#define C_TGT   2048
#define C_MEM   512
#define C_SRC   2560   // MEM + TGT
#define C_B     4
#define C_INF   2048
#define C_PROJ  1024
#define C_OUTF  2048
#define C_HEADS 8
#define C_HD    128

typedef __attribute__((ext_vector_type(16))) __bf16 v16bf;
typedef __attribute__((ext_vector_type(8)))  float  v8f;

union Frag16 { unsigned int u[8]; v16bf v; };

__device__ __forceinline__ unsigned short f2bf(float f) {
    unsigned int u = __float_as_uint(f);
    u += 0x7FFFu + ((u >> 16) & 1u);   // round-to-nearest-even
    return (unsigned short)(u >> 16);
}

// ---------------- gfx1250 async global->LDS copy (guarded) ----------------
#if defined(__HIP_DEVICE_COMPILE__) && \
    __has_builtin(__builtin_amdgcn_global_load_async_to_lds_b128) && \
    __has_builtin(__builtin_amdgcn_s_wait_asynccnt)
#define USE_ASYNC 1
#else
#define USE_ASYNC 0
#endif

// 16-byte payload type matching the builtin's expected pointer element type
typedef int b128i __attribute__((vector_size(16)));
typedef __attribute__((address_space(1))) b128i* gbl_b128_p;
typedef __attribute__((address_space(3))) b128i* lds_b128_p;

// copy 16 bytes global -> LDS. Both addresses must be 16B aligned.
__device__ __forceinline__ void copy16(const void* g, void* l) {
#if USE_ASYNC
    __builtin_amdgcn_global_load_async_to_lds_b128(
        (gbl_b128_p)(uintptr_t)g,
        (lds_b128_p)(unsigned int)(uintptr_t)l,
        0, 0);
#else
    *(uint4*)l = *(const uint4*)g;
#endif
}

__device__ __forceinline__ void wait_async_copies() {
#if USE_ASYNC
    __builtin_amdgcn_s_wait_asynccnt(0);
#endif
}

// ---------------- fp32 -> bf16 convert ----------------
__global__ void cvt_f32_bf16(const float* __restrict__ src,
                             unsigned short* __restrict__ dst, int n) {
    int i = blockIdx.x * 256 + threadIdx.x;
    if (i < n) dst[i] = f2bf(src[i]);
}

// ---------------- generic bf16 WMMA GEMM: C = A(MxK) * B(NxK)^T ----------------
// block = 128 threads (4 waves), block tile 128x64, per-wave 32x64 (8 WMMA/K-step)
// double-buffered LDS, K-step 32; row stride 40 halves (80B, 16B-aligned chunks)
__global__ __launch_bounds__(128) void gemm_bf16(
    const unsigned short* __restrict__ A,
    const unsigned short* __restrict__ B,
    float* __restrict__ Cf,            // optional fp32 output
    unsigned short* __restrict__ Cb,   // optional bf16 output
    int M, int N, int K)
{
    __shared__ unsigned short As[2][128][40];
    __shared__ unsigned short Bs[2][64][40];

    const int tid  = threadIdx.x;
    const int lane = tid & 31;
    const int wid  = tid >> 5;
    const int hid  = lane >> 4;
    const int l15  = lane & 15;
    const int mbase = blockIdx.y * 128;
    const int nbase = blockIdx.x * 64;
    const int k0 = hid * 8;

    v8f acc[2][4];
    for (int mt = 0; mt < 2; ++mt)
        for (int nt = 0; nt < 4; ++nt)
            for (int r = 0; r < 8; ++r) acc[mt][nt][r] = 0.0f;

    const int nkb = K >> 5;

    // tile loader: A 128x32, B 64x32 (halves), 16B chunks
    auto load_tiles = [&](int kb, int buf) {
        #pragma unroll
        for (int i = 0; i < 4; ++i) {
            int idx = tid + i * 128;
            int r = idx >> 2, c = (idx & 3) * 8;
            copy16(A + (size_t)(mbase + r) * K + kb + c, &As[buf][r][c]);
        }
        #pragma unroll
        for (int i = 0; i < 2; ++i) {
            int idx = tid + i * 128;
            int r = idx >> 2, c = (idx & 3) * 8;
            copy16(B + (size_t)(nbase + r) * K + kb + c, &Bs[buf][r][c]);
        }
    };

    load_tiles(0, 0);
    wait_async_copies();
    __syncthreads();

    for (int ki = 0; ki < nkb; ++ki) {
        int cur = ki & 1;
        if (ki + 1 < nkb) load_tiles((ki + 1) << 5, cur ^ 1);

        // two A fragments (rows wid*32 .. wid*32+31)
        Frag16 af[2];
        #pragma unroll
        for (int mt = 0; mt < 2; ++mt) {
            int arow = wid * 32 + mt * 16 + l15;
            #pragma unroll
            for (int j = 0; j < 4; ++j)
                af[mt].u[j]     = *(const unsigned int*)&As[cur][arow][k0 + 2 * j];
            #pragma unroll
            for (int j = 0; j < 4; ++j)
                af[mt].u[4 + j] = *(const unsigned int*)&As[cur][arow][16 + k0 + 2 * j];
        }
        #pragma unroll
        for (int nt = 0; nt < 4; ++nt) {
            Frag16 bf;
            int ncol = nt * 16 + l15;
            #pragma unroll
            for (int j = 0; j < 8; ++j)
                bf.u[j] = *(const unsigned int*)&Bs[cur][ncol][hid * 8 + 2 * j + (j >= 4 ? 8 : 0)];
            acc[0][nt] = __builtin_amdgcn_wmma_f32_16x16x32_bf16(
                false, af[0].v, false, bf.v, (short)0, acc[0][nt], false, false);
            acc[1][nt] = __builtin_amdgcn_wmma_f32_16x16x32_bf16(
                false, af[1].v, false, bf.v, (short)0, acc[1][nt], false, false);
        }
        wait_async_copies();
        __syncthreads();
    }

    // writeback: element r -> row (r + 8*hid), col l15 within tile
    for (int mt = 0; mt < 2; ++mt) {
        for (int nt = 0; nt < 4; ++nt) {
            int col = nbase + nt * 16 + l15;
            #pragma unroll
            for (int r = 0; r < 8; ++r) {
                int row = mbase + wid * 32 + mt * 16 + hid * 8 + r;
                float v = acc[mt][nt][r];
                if (Cf) Cf[(size_t)row * N + col] = v;
                if (Cb) Cb[(size_t)row * N + col] = f2bf(v);
            }
        }
    }
}

// ---------------- flash attention ----------------
// grid: (TGT/128, B, HEADS); block = 256 threads (8 waves, each owns 16 q rows)
__global__ __launch_bounds__(256) void attn_kernel(
    const unsigned short* __restrict__ Zbf,   // (SRC*B, PROJ) bf16  (q source z)
    const unsigned short* __restrict__ KVbf,  // (SRC*B, 2*PROJ) bf16
    const float* __restrict__ attn_mask,      // (TGT, SRC)
    const unsigned char* __restrict__ mask_pad,      // (TGT, B)
    const unsigned char* __restrict__ mem_mask_pad,  // (MEM, B)
    float* __restrict__ AO)                   // (TGT*B, PROJ) fp32
{
    __shared__ unsigned short Ks[32][136];    // K chunk (16B-aligned rows)
    __shared__ unsigned short Vt[128][36];    // V chunk transposed: [d][s-local]
    __shared__ unsigned short Ps[8][16][34];  // per-wave P tile (16 x 32)

    const int tid  = threadIdx.x;
    const int lane = tid & 31;
    const int wid  = tid >> 5;
    const int hid  = lane >> 4;
    const int l15  = lane & 15;
    const int tb   = blockIdx.x * 128;
    const int b    = blockIdx.y;
    const int h    = blockIdx.z;
    const float rscale = 0.08838834764831845f;  // 128^-0.5

    // Q fragments (16 rows x 128 d, as 4 A-fragments of K=32), loaded once
    Frag16 qf[4];
    {
        int trow = tb + wid * 16 + l15;
        size_t base = (size_t)((C_MEM + trow) * C_B + b) * C_PROJ + h * C_HD;
        int k0 = hid * 8;
        #pragma unroll
        for (int kc = 0; kc < 4; ++kc) {
            #pragma unroll
            for (int j = 0; j < 4; ++j)
                qf[kc].u[j]     = *(const unsigned int*)(Zbf + base + kc * 32 + k0 + 2 * j);
            #pragma unroll
            for (int j = 0; j < 4; ++j)
                qf[kc].u[4 + j] = *(const unsigned int*)(Zbf + base + kc * 32 + 16 + k0 + 2 * j);
        }
    }

    v8f o[8];
    float mrow[8], lrow[8];
    #pragma unroll
    for (int nt = 0; nt < 8; ++nt)
        for (int r = 0; r < 8; ++r) o[nt][r] = 0.0f;
    #pragma unroll
    for (int r = 0; r < 8; ++r) { mrow[r] = -3.0e38f; lrow[r] = 0.0f; }

    int s_end = tb + 128 + C_MEM;
    if (s_end > C_SRC) s_end = C_SRC;

    for (int sb = 0; sb < s_end; sb += 32) {
        // --- stage K chunk (32 x 128), async when available ---
        for (int i = 0; i < 2; ++i) {
            int idx = tid + i * 256;
            int r = idx >> 4, c = (idx & 15) * 8;
            copy16(KVbf + (size_t)((sb + r) * C_B + b) * (2 * C_PROJ) + h * C_HD + c,
                   &Ks[r][c]);
        }
        // --- stage V chunk transposed (d x 32) ---
        {
            int sl = tid >> 3;
            int dbase = (tid & 7) * 16;
            const unsigned short* vp = KVbf + (size_t)((sb + sl) * C_B + b) * (2 * C_PROJ)
                                       + C_PROJ + h * C_HD + dbase;
            #pragma unroll
            for (int j = 0; j < 16; ++j) Vt[dbase + j][sl] = vp[j];
        }
        wait_async_copies();
        __syncthreads();

        // --- scores + online softmax for two 16-col tiles ---
        #pragma unroll
        for (int nt2 = 0; nt2 < 2; ++nt2) {
            v8f sc;
            for (int r = 0; r < 8; ++r) sc[r] = 0.0f;
            #pragma unroll
            for (int kc = 0; kc < 4; ++kc) {
                Frag16 kf;
                int srow = nt2 * 16 + l15;
                #pragma unroll
                for (int j = 0; j < 8; ++j)
                    kf.u[j] = *(const unsigned int*)&Ks[srow][kc * 32 + hid * 8 + 2 * j + (j >= 4 ? 8 : 0)];
                sc = __builtin_amdgcn_wmma_f32_16x16x32_bf16(
                    false, qf[kc].v, false, kf.v, (short)0, sc, false, false);
            }
            int s = sb + nt2 * 16 + l15;
            unsigned char pv = (s < C_MEM) ? mem_mask_pad[s * C_B + b]
                                           : mask_pad[(s - C_MEM) * C_B + b];
            #pragma unroll
            for (int r = 0; r < 8; ++r) {
                int t = tb + wid * 16 + hid * 8 + r;
                float v = sc[r] * rscale + attn_mask[(size_t)t * C_SRC + s];
                if (pv) v = -1.0e30f;
                sc[r] = v;
            }
            // online softmax per row (row stats replicated across its 16 lanes)
            #pragma unroll
            for (int r = 0; r < 8; ++r) {
                float v = sc[r];
                float mt = v;
                mt = fmaxf(mt, __shfl_xor(mt, 1, 32));
                mt = fmaxf(mt, __shfl_xor(mt, 2, 32));
                mt = fmaxf(mt, __shfl_xor(mt, 4, 32));
                mt = fmaxf(mt, __shfl_xor(mt, 8, 32));
                float mn = fmaxf(mrow[r], mt);
                float p  = __expf(v - mn);
                float pt = p;
                pt += __shfl_xor(pt, 1, 32);
                pt += __shfl_xor(pt, 2, 32);
                pt += __shfl_xor(pt, 4, 32);
                pt += __shfl_xor(pt, 8, 32);
                float alpha = __expf(mrow[r] - mn);
                lrow[r] = lrow[r] * alpha + pt;
                mrow[r] = mn;
                #pragma unroll
                for (int nt = 0; nt < 8; ++nt) o[nt][r] *= alpha;
                Ps[wid][hid * 8 + r][nt2 * 16 + l15] = f2bf(p);
            }
        }

        // --- O += P (16x32) @ V (32x128) ---  (Ps is wave-private; in-wave LDS order)
        {
            Frag16 pf;
            int k0 = hid * 8;
            #pragma unroll
            for (int j = 0; j < 4; ++j)
                pf.u[j]     = *(const unsigned int*)&Ps[wid][l15][k0 + 2 * j];
            #pragma unroll
            for (int j = 0; j < 4; ++j)
                pf.u[4 + j] = *(const unsigned int*)&Ps[wid][l15][16 + k0 + 2 * j];
            #pragma unroll
            for (int nt = 0; nt < 8; ++nt) {
                Frag16 vf;
                int dcol = nt * 16 + l15;
                #pragma unroll
                for (int j = 0; j < 8; ++j)
                    vf.u[j] = *(const unsigned int*)&Vt[dcol][hid * 8 + 2 * j + (j >= 4 ? 8 : 0)];
                o[nt] = __builtin_amdgcn_wmma_f32_16x16x32_bf16(
                    false, pf.v, false, vf.v, (short)0, o[nt], false, false);
            }
        }
        __syncthreads();
    }

    // normalize and write
    #pragma unroll
    for (int nt = 0; nt < 8; ++nt) {
        int col = h * C_HD + nt * 16 + l15;
        #pragma unroll
        for (int r = 0; r < 8; ++r) {
            int t = tb + wid * 16 + hid * 8 + r;
            AO[(size_t)(t * C_B + b) * C_PROJ + col] = o[nt][r] / lrow[r];
        }
    }
}

// ---------------- residual add + LayerNorm -> bf16 ----------------
__global__ __launch_bounds__(256) void ln_kernel(
    const float* __restrict__ AO,     // (TGT*B, PROJ)
    const float* __restrict__ Zf,     // (SRC*B, PROJ), residual at rows + MEM*B
    const float* __restrict__ gamma,
    const float* __restrict__ beta,
    unsigned short* __restrict__ out) // (TGT*B, PROJ) bf16
{
    __shared__ float s1[256], s2[256];
    const int row = blockIdx.x;
    const int tid = threadIdx.x;
    float v[4], sum = 0.0f, sq = 0.0f;
    #pragma unroll
    for (int j = 0; j < 4; ++j) {
        int c = tid + j * 256;
        float x = AO[(size_t)row * C_PROJ + c]
                + Zf[(size_t)(row + C_MEM * C_B) * C_PROJ + c];
        v[j] = x; sum += x; sq += x * x;
    }
    s1[tid] = sum; s2[tid] = sq;
    __syncthreads();
    for (int off = 128; off > 0; off >>= 1) {
        if (tid < off) { s1[tid] += s1[tid + off]; s2[tid] += s2[tid + off]; }
        __syncthreads();
    }
    float mean = s1[0] * (1.0f / C_PROJ);
    float var  = s2[0] * (1.0f / C_PROJ) - mean * mean;
    float rstd = rsqrtf(var + 1e-5f);
    #pragma unroll
    for (int j = 0; j < 4; ++j) {
        int c = tid + j * 256;
        float y = (v[j] - mean) * rstd * gamma[c] + beta[c];
        out[(size_t)row * C_PROJ + c] = f2bf(y);
    }
}

// ---------------- host launcher ----------------
extern "C" void kernel_launch(void* const* d_in, const int* in_sizes, int n_in,
                              void* d_out, int out_size, void* d_ws, size_t ws_size,
                              hipStream_t stream) {
    const float* x       = (const float*)d_in[0];  // (2048,4,2048)
    const float* memory  = (const float*)d_in[1];  // (512,4,2048)
    const float* W1      = (const float*)d_in[2];  // (1024,2048)
    const float* W2      = (const float*)d_in[3];  // (2048,1024)
    const float* W3      = (const float*)d_in[4];  // (2048,1024)
    const float* gamma   = (const float*)d_in[5];
    const float* beta    = (const float*)d_in[6];
    const float* attn_mask = (const float*)d_in[7];        // (2048,2560)
    const unsigned char* mask_pad     = (const unsigned char*)d_in[8];
    const unsigned char* mem_mask_pad = (const unsigned char*)d_in[9];
    float* out = (float*)d_out;

    const size_t ROWS  = (size_t)C_SRC * C_B;   // 10240
    const size_t TROWS = (size_t)C_TGT * C_B;   // 8192

    char* w = (char*)d_ws;
    auto alloc = [&](size_t bytes) -> char* {
        char* p = w; w += (bytes + 255) & ~(size_t)255; return p;
    };
    unsigned short* Abf  = (unsigned short*)alloc(ROWS * C_INF  * 2);  // concat input bf16
    unsigned short* W1bf = (unsigned short*)alloc((size_t)C_PROJ * C_INF * 2);
    unsigned short* W2bf = (unsigned short*)alloc((size_t)2 * C_PROJ * C_PROJ * 2);
    unsigned short* W3bf = (unsigned short*)alloc((size_t)C_OUTF * C_PROJ * 2);
    float*          Zf   = (float*)alloc(ROWS * C_PROJ * 4);
    unsigned short* Zbf  = (unsigned short*)alloc(ROWS * C_PROJ * 2);
    unsigned short* KVbf = (unsigned short*)alloc(ROWS * 2 * C_PROJ * 2);
    float*          AO   = (float*)alloc(TROWS * C_PROJ * 4);
    unsigned short* LNbf = (unsigned short*)alloc(TROWS * C_PROJ * 2);

    // convert inputs to bf16 (memory rows first, then x rows -> concat)
    {
        int nmem = C_MEM * C_B * C_INF;          // 4194304
        int nx   = C_TGT * C_B * C_INF;          // 16777216
        cvt_f32_bf16<<<(nmem + 255) / 256, 256, 0, stream>>>(memory, Abf, nmem);
        cvt_f32_bf16<<<(nx   + 255) / 256, 256, 0, stream>>>(x, Abf + nmem, nx);
        int nw1 = C_PROJ * C_INF;
        int nw2 = 2 * C_PROJ * C_PROJ;
        int nw3 = C_OUTF * C_PROJ;
        cvt_f32_bf16<<<(nw1 + 255) / 256, 256, 0, stream>>>(W1, W1bf, nw1);
        cvt_f32_bf16<<<(nw2 + 255) / 256, 256, 0, stream>>>(W2, W2bf, nw2);
        cvt_f32_bf16<<<(nw3 + 255) / 256, 256, 0, stream>>>(W3, W3bf, nw3);
    }

    // GEMM1: Z = Abf (10240x2048) @ W1^T -> fp32 + bf16
    gemm_bf16<<<dim3(C_PROJ / 64, (int)(ROWS / 128)), 128, 0, stream>>>(
        Abf, W1bf, Zf, Zbf, (int)ROWS, C_PROJ, C_INF);

    // GEMM2: KV = Zbf (10240x1024) @ W2^T -> bf16 only
    gemm_bf16<<<dim3(2 * C_PROJ / 64, (int)(ROWS / 128)), 128, 0, stream>>>(
        Zbf, W2bf, nullptr, KVbf, (int)ROWS, 2 * C_PROJ, C_PROJ);

    // Flash attention
    attn_kernel<<<dim3(C_TGT / 128, C_B, C_HEADS), 256, 0, stream>>>(
        Zbf, KVbf, attn_mask, mask_pad, mem_mask_pad, AO);

    // residual + LayerNorm -> bf16
    ln_kernel<<<(int)TROWS, 256, 0, stream>>>(AO, Zf, gamma, beta, LNbf);

    // GEMM3: out = LNbf (8192x1024) @ W3^T -> fp32 d_out
    gemm_bf16<<<dim3(C_OUTF / 64, (int)(TROWS / 128)), 128, 0, stream>>>(
        LNbf, W3bf, out, nullptr, (int)TROWS, C_OUTF, C_PROJ);
}
